// GatedCrossAttention_53901839565376
// MI455X (gfx1250) — matrix operand order repaired and checked by reading, not compiled, not run on hardware
//
#include <hip/hip_runtime.h>
#include <hip/hip_bf16.h>
#include <math.h>

// ---------------- CDNA5 WMMA types ----------------
typedef __attribute__((ext_vector_type(16))) __bf16 v16bf;
typedef __attribute__((ext_vector_type(8)))  float  v8f;

#define D_MODEL  2560
#define N_HEADS  32
#define HEAD_DIM 80
#define DH_PAD   96           // pad head dim to a multiple of 32 for the K loop
#define SEQ_Q    2048
#define SEQ_KV   512
#define BATCH    2
#define QSCALE   0.11180339887498949f   // 1/sqrt(80)

union FragBF {
  v16bf  v;
  uint4  u[2];
  __bf16 h[16];
};

// A-matrix 16x32 bf16 layout (ISA 7.12.2): lane L holds row M=L%16.
// lanes 0-15: K = [kk+0,kk+8) and [kk+16,kk+24); lanes 16-31: +8.
__device__ __forceinline__ v16bf load_a_frag(const __bf16* base, int ld, int lane, int kk) {
  const __bf16* p = base + (size_t)(lane & 15) * ld + kk + ((lane >> 4) << 3);
  FragBF f;
  f.u[0] = *reinterpret_cast<const uint4*>(p);
  f.u[1] = *reinterpret_cast<const uint4*>(p + 16);
  return f.v;
}

// B-matrix 32x16 bf16: lane L holds column N=L%16, same K order as A:
// lanes 0-15: K = [kk+0,kk+8) and [kk+16,kk+24); lanes 16-31: +8 (row of W, C = X*W^T).
__device__ __forceinline__ v16bf load_b_frag(const __bf16* base, int ld, int lane, int kk) {
  const __bf16* p = base + (size_t)(lane & 15) * ld + kk + ((lane >> 4) << 3);
  FragBF f;
  f.u[0] = *reinterpret_cast<const uint4*>(p);
  f.u[1] = *reinterpret_cast<const uint4*>(p + 16);
  return f.v;
}

__device__ __forceinline__ v8f wmma_bf16(v16bf a, v16bf b, v8f c) {
  return __builtin_amdgcn_wmma_f32_16x16x32_bf16(false, a, false, b, (short)0, c, false, false);
}

// ---------------- cast f32 -> bf16 ----------------
__global__ void cast_f32_to_bf16(const float* __restrict__ in, __bf16* __restrict__ out, size_t n) {
  size_t i      = (size_t)blockIdx.x * blockDim.x + threadIdx.x;
  size_t stride = (size_t)gridDim.x * blockDim.x;
  for (; i < n; i += stride) out[i] = (__bf16)in[i];
}

// ---------------- QKV projection GEMM ----------------
// C[M, 2560] = X[M, 2560] * W^T, W stored [out=2560, in=2560] (nn.Linear).
// Each wave computes a 32x64 tile: 2 A-frags x 4 B-frags -> 8 WMMAs per K-step
// (12 b128 loads), halving operand refetch vs a 16x64 tile.
// MODE 0: Q -> [B,H,S,DH_PAD] * QSCALE   MODE 1: K -> [B,H,L,DH_PAD]
// MODE 2: V -> transposed [B,H,HEAD_DIM,L]
template <int MODE>
__global__ void gemm_qkv(const __bf16* __restrict__ X, const __bf16* __restrict__ W,
                         int M, int rowsPerBatch,
                         __bf16* __restrict__ Q, __bf16* __restrict__ KT, __bf16* __restrict__ VT) {
  const int lane = threadIdx.x & 31;
  const int wave = threadIdx.x >> 5;
  const int ntN  = D_MODEL / 64;                       // 40 N-tiles of 64
  int wid   = blockIdx.x * (blockDim.x >> 5) + wave;
  int total = (M >> 5) * ntN;
  if (wid >= total) return;
  int mt = wid / ntN, nt = wid % ntN;
  int m0 = mt << 5, n0 = nt << 6;

  v8f acc[2][4] = {};
  const __bf16* Xb0 = X + (size_t)m0 * D_MODEL;
  const __bf16* Xb1 = Xb0 + (size_t)16 * D_MODEL;
  for (int kk = 0; kk < D_MODEL; kk += 32) {
    v16bf a0 = load_a_frag(Xb0, D_MODEL, lane, kk);
    v16bf a1 = load_a_frag(Xb1, D_MODEL, lane, kk);
    // pull next K-block's lines into WGP$ (global_prefetch_b8)
    __builtin_prefetch(Xb0 + (size_t)(lane & 15) * D_MODEL + kk + 128, 0, 0);
    __builtin_prefetch(Xb1 + (size_t)(lane & 15) * D_MODEL + kk + 128, 0, 0);
#pragma unroll
    for (int j = 0; j < 4; ++j) {
      const __bf16* Wr = W + (size_t)(n0 + 16 * j) * D_MODEL;
      v16bf b = load_b_frag(Wr, D_MODEL, lane, kk);
      __builtin_prefetch(Wr + (size_t)(lane & 15) * D_MODEL + kk + 128, 0, 0);
      acc[0][j] = wmma_bf16(a0, b, acc[0][j]);
      acc[1][j] = wmma_bf16(a1, b, acc[1][j]);
    }
  }

  const int lrow = lane >> 4, lcol = lane & 15;
#pragma unroll
  for (int i = 0; i < 2; ++i) {
#pragma unroll
    for (int j = 0; j < 4; ++j) {
      int nb = n0 + 16 * j;
      int h  = nb / HEAD_DIM;               // 80 % 16 == 0 -> tile never straddles a head
      int d  = (nb % HEAD_DIM) + lcol;
#pragma unroll
      for (int r = 0; r < 8; ++r) {
        int   m  = m0 + 16 * i + r + (lrow << 3);
        int   b  = m / rowsPerBatch;
        int   sl = m % rowsPerBatch;
        float val = acc[i][j][r];
        if (MODE == 0)
          Q[((size_t)(b * N_HEADS + h) * rowsPerBatch + sl) * DH_PAD + d] = (__bf16)(val * QSCALE);
        else if (MODE == 1)
          KT[((size_t)(b * N_HEADS + h) * rowsPerBatch + sl) * DH_PAD + d] = (__bf16)val;
        else
          VT[((size_t)(b * N_HEADS + h) * HEAD_DIM + d) * SEQ_KV + sl] = (__bf16)val;
      }
    }
  }
}

// ---------------- fused attention: scores -> softmax -> weights out -> A*V ----------------
// One wave owns 16 query rows x 512 keys; score strip lives in a private 32KB LDS slice.
__global__ void attn_kernel(const __bf16* __restrict__ Q, const __bf16* __restrict__ KT,
                            const __bf16* __restrict__ VT,
                            float* __restrict__ attn_w, __bf16* __restrict__ AO) {
  extern __shared__ float smem[];                      // wavesPerBlock * 16 * 512 floats
  const int lane = threadIdx.x & 31;
  const int wave = threadIdx.x >> 5;
  float* sc = smem + (size_t)wave * 16 * SEQ_KV;

  int tid = blockIdx.x * (blockDim.x >> 5) + wave;     // 0 .. B*H*(S/16)-1 = 8191
  int bh  = tid >> 7;                                  // / (S/16)
  int s0  = (tid & 127) << 4;
  const int lrow = lane >> 4, lcol = lane & 15;

  // ---- scores = (Q * scale) @ K^T : hoist the 3 Q A-frags (K = 96 = 3*32)
  const __bf16* Qb = Q + ((size_t)bh * SEQ_Q + s0) * DH_PAD;
  v16bf aq[3];
#pragma unroll
  for (int i = 0; i < 3; ++i) aq[i] = load_a_frag(Qb, DH_PAD, lane, 32 * i);

  const __bf16* Kb = KT + (size_t)bh * SEQ_KV * DH_PAD;
  for (int lt = 0; lt < SEQ_KV / 16; ++lt) {
    int l0  = lt << 4;
    v8f acc = {};
#pragma unroll
    for (int i = 0; i < 3; ++i) {
      v16bf bf = load_b_frag(Kb + (size_t)l0 * DH_PAD, DH_PAD, lane, 32 * i);
      acc = wmma_bf16(aq[i], bf, acc);
    }
#pragma unroll
    for (int r = 0; r < 8; ++r)
      sc[(r + (lrow << 3)) * SEQ_KV + l0 + lcol] = acc[r];
  }

  // ---- softmax over L=512 per row; lane pair (l, l+16) splits a row in halves of 256
  const int row  = lane & 15;
  const int half = lane >> 4;
  float* rp = sc + row * SEQ_KV + half * 256;
  float mx = -3.4e38f;
  for (int c = 0; c < 256; ++c) mx = fmaxf(mx, rp[c]);
  mx = fmaxf(mx, __shfl_xor(mx, 16, 32));
  float sum = 0.f;
  for (int c = 0; c < 256; ++c) { float e = __expf(rp[c] - mx); rp[c] = e; sum += e; }
  sum += __shfl_xor(sum, 16, 32);
  float inv = 1.0f / sum;

  // normalize in LDS + emit attn_weights (f32, second output)
  float* wout = attn_w + ((size_t)bh * SEQ_Q + s0 + row) * SEQ_KV + half * 256;
  for (int c = 0; c < 256; c += 4) {
    float4 v = *reinterpret_cast<float4*>(rp + c);
    v.x *= inv; v.y *= inv; v.z *= inv; v.w *= inv;
    *reinterpret_cast<float4*>(rp + c)   = v;
    *reinterpret_cast<float4*>(wout + c) = v;
  }

  // ---- attn_out = weights @ V : A from LDS (f32->bf16), B = rows of V^T [Dh, L]
  v8f oacc[5] = {};
  const __bf16* Vb = VT + (size_t)bh * HEAD_DIM * SEQ_KV;
  for (int ks = 0; ks < SEQ_KV / 32; ++ks) {
    int kb = ks << 5;
    FragBF fa;
    const float* ap = sc + (size_t)(lane & 15) * SEQ_KV + kb + ((lane >> 4) << 3);
#pragma unroll
    for (int i = 0; i < 8; ++i) fa.h[i]     = (__bf16)ap[i];
#pragma unroll
    for (int i = 0; i < 8; ++i) fa.h[8 + i] = (__bf16)ap[16 + i];
#pragma unroll
    for (int ntl = 0; ntl < 5; ++ntl) {
      v16bf bf = load_b_frag(Vb + (size_t)(ntl * 16) * SEQ_KV, SEQ_KV, lane, kb);
      oacc[ntl] = wmma_bf16(fa.v, bf, oacc[ntl]);
    }
  }

  // store AO[b, s, h*80+d] (bf16, heads re-interleaved for the output projection)
  int b = bh >> 5, h = bh & 31;
#pragma unroll
  for (int ntl = 0; ntl < 5; ++ntl) {
    int d = ntl * 16 + lcol;
#pragma unroll
    for (int r = 0; r < 8; ++r) {
      int m = r + (lrow << 3);
      AO[((size_t)b * SEQ_Q + s0 + m) * D_MODEL + h * HEAD_DIM + d] = (__bf16)oacc[ntl][r];
    }
  }
}

// ---------------- output projection with fused tanh gate (32x64 wave tile) ----------------
__global__ void gemm_out(const __bf16* __restrict__ AO, const __bf16* __restrict__ Wo,
                         const float* __restrict__ gate_p, float* __restrict__ out, int M) {
  const int lane = threadIdx.x & 31;
  const int wave = threadIdx.x >> 5;
  const int ntN  = D_MODEL / 64;
  int wid   = blockIdx.x * (blockDim.x >> 5) + wave;
  int total = (M >> 5) * ntN;
  if (wid >= total) return;
  int mt = wid / ntN, nt = wid % ntN;
  int m0 = mt << 5, n0 = nt << 6;

  v8f acc[2][4] = {};
  const __bf16* Xb0 = AO + (size_t)m0 * D_MODEL;
  const __bf16* Xb1 = Xb0 + (size_t)16 * D_MODEL;
  for (int kk = 0; kk < D_MODEL; kk += 32) {
    v16bf a0 = load_a_frag(Xb0, D_MODEL, lane, kk);
    v16bf a1 = load_a_frag(Xb1, D_MODEL, lane, kk);
    __builtin_prefetch(Xb0 + (size_t)(lane & 15) * D_MODEL + kk + 128, 0, 0);
    __builtin_prefetch(Xb1 + (size_t)(lane & 15) * D_MODEL + kk + 128, 0, 0);
#pragma unroll
    for (int j = 0; j < 4; ++j) {
      const __bf16* Wr = Wo + (size_t)(n0 + 16 * j) * D_MODEL;
      v16bf b = load_b_frag(Wr, D_MODEL, lane, kk);
      __builtin_prefetch(Wr + (size_t)(lane & 15) * D_MODEL + kk + 128, 0, 0);
      acc[0][j] = wmma_bf16(a0, b, acc[0][j]);
      acc[1][j] = wmma_bf16(a1, b, acc[1][j]);
    }
  }

  float gate = tanhf(gate_p[0]);
  const int lrow = lane >> 4, lcol = lane & 15;
#pragma unroll
  for (int i = 0; i < 2; ++i) {
#pragma unroll
    for (int j = 0; j < 4; ++j) {
#pragma unroll
      for (int r = 0; r < 8; ++r) {
        int m = m0 + 16 * i + r + (lrow << 3);
        out[(size_t)m * D_MODEL + n0 + 16 * j + lcol] = gate * acc[i][j][r];
      }
    }
  }
}

// ---------------- host launcher ----------------
extern "C" void kernel_launch(void* const* d_in, const int* in_sizes, int n_in,
                              void* d_out, int out_size, void* d_ws, size_t ws_size,
                              hipStream_t stream) {
  const float* hs   = (const float*)d_in[0];
  const float* ls   = (const float*)d_in[1];
  const float* Wq   = (const float*)d_in[2];
  const float* Wk   = (const float*)d_in[3];
  const float* Wv   = (const float*)d_in[4];
  const float* Wo   = (const float*)d_in[5];
  const float* gate = (const float*)d_in[6];
  float* out = (float*)d_out;

  const size_t nHS = (size_t)BATCH * SEQ_Q  * D_MODEL;   // 10,485,760
  const size_t nLS = (size_t)BATCH * SEQ_KV * D_MODEL;   //  2,621,440
  const size_t nW  = (size_t)D_MODEL * D_MODEL;          //  6,553,600

  // carve workspace (bf16 buffers), 256B aligned
  char*  ws  = (char*)d_ws;
  size_t off = 0;
  auto carve = [&](size_t bytes) -> char* {
    char* p = ws + off;
    off += (bytes + 255) & ~(size_t)255;
    return p;
  };
  __bf16* XH  = (__bf16*)carve(nHS * 2);
  __bf16* XL  = (__bf16*)carve(nLS * 2);
  __bf16* WQb = (__bf16*)carve(nW * 2);
  __bf16* WKb = (__bf16*)carve(nW * 2);
  __bf16* WVb = (__bf16*)carve(nW * 2);
  __bf16* WOb = (__bf16*)carve(nW * 2);
  const size_t qBytes  = (size_t)BATCH * N_HEADS * SEQ_Q  * DH_PAD * 2;   // 25,165,824
  const size_t ktBytes = (size_t)BATCH * N_HEADS * SEQ_KV * DH_PAD * 2;   //  6,291,456
  const size_t vtBytes = (size_t)BATCH * N_HEADS * HEAD_DIM * SEQ_KV * 2; //  5,242,880
  __bf16* Qb  = (__bf16*)carve(qBytes);
  __bf16* KTb = (__bf16*)carve(ktBytes);
  __bf16* VTb = (__bf16*)carve(vtBytes);
  __bf16* AOb = (__bf16*)carve(nHS * 2);
  (void)ws_size; (void)in_sizes; (void)n_in; (void)out_size;

  // 1) casts to bf16
  cast_f32_to_bf16<<<4096, 256, 0, stream>>>(hs, XH, nHS);
  cast_f32_to_bf16<<<4096, 256, 0, stream>>>(ls, XL, nLS);
  cast_f32_to_bf16<<<4096, 256, 0, stream>>>(Wq, WQb, nW);
  cast_f32_to_bf16<<<4096, 256, 0, stream>>>(Wk, WKb, nW);
  cast_f32_to_bf16<<<4096, 256, 0, stream>>>(Wv, WVb, nW);
  cast_f32_to_bf16<<<4096, 256, 0, stream>>>(Wo, WOb, nW);

  // 2) zero Q/K pad columns (d in [80,96)) by clearing the buffers
  hipMemsetAsync(Qb, 0, qBytes, stream);
  hipMemsetAsync(KTb, 0, ktBytes, stream);

  // 3) QKV projections: waves = (M/32)*(2560/64); 8 waves / 256-thread block
  gemm_qkv<0><<<640, 256, 0, stream>>>(XH, WQb, BATCH * SEQ_Q,  SEQ_Q,  Qb, KTb, VTb);
  gemm_qkv<1><<<160, 256, 0, stream>>>(XL, WKb, BATCH * SEQ_KV, SEQ_KV, Qb, KTb, VTb);
  gemm_qkv<2><<<160, 256, 0, stream>>>(XL, WVb, BATCH * SEQ_KV, SEQ_KV, Qb, KTb, VTb);

  // 4) fused attention: 8192 wave-tiles, 4 waves/block, 128KB dynamic LDS
  float* attnW = out + nHS;  // second output: [B,H,S,L] f32
  attn_kernel<<<2048, 128, 4 * 16 * SEQ_KV * sizeof(float), stream>>>(Qb, KTb, VTb, attnW, AOb);

  // 5) output projection with tanh(gate) fused
  gemm_out<<<640, 256, 0, stream>>>(AOb, WOb, gate, out, BATCH * SEQ_Q);
}
